// fusion_loss_2791728742759
// MI455X (gfx1250) — matrix operand order repaired and checked by reading, not compiled, not run on hardware
//
#include <hip/hip_runtime.h>
#include <math.h>

typedef __attribute__((ext_vector_type(16))) _Float16 v16h;
typedef __attribute__((ext_vector_type(8)))  _Float16 v8h;
typedef __attribute__((ext_vector_type(2)))  _Float16 v2h;
typedef __attribute__((ext_vector_type(8)))  float    v8f;

#define B_    8
#define H_    512
#define W_    512
#define TILE  32
#define HALO  5
#define RT    (TILE + 2*HALO)   // 42 raw tile rows/cols (halo 5 covers sobel=1, box=4, gauss=5)
#define RP    (RT + 1)          // 43 pitch, odd -> conflict-free column access
#define BR    (TILE + 8)        // 40 box-horizontal temp rows (vertical halo 4)
#define BP    (TILE + 1)        // 33
#define NT    128               // 4 wave32 per block
#define KP    40                // ssim scratch pitch in halves: 80B/col, 16B-aligned segs
#define SCRF  6400              // scratch floats: max(3*BR*BP=3960, 4 waves*5ch*16*40/2=6400)

// 11-tap gaussian, sigma=1.5, normalized
__constant__ float G11[11] = {
    0.0010284f, 0.0075988f, 0.0360008f, 0.1093607f, 0.2130056f,
    0.2660117f,
    0.2130056f, 0.1093607f, 0.0360008f, 0.0075988f, 0.0010284f };

__device__ __forceinline__ float sobel_at(const float* S, int o) {
  const float a00=S[o],      a01=S[o+1],      a02=S[o+2];
  const float a10=S[o+RP],                    a12=S[o+RP+2];
  const float a20=S[o+2*RP], a21=S[o+2*RP+1], a22=S[o+2*RP+2];
  const float gx = (a02 - a00) + 2.f*(a12 - a10) + (a22 - a20);
  const float gy = (a00 - a20) + 2.f*(a01 - a21) + (a02 - a22);
  return fabsf(gx) + fabsf(gy);
}

__device__ __forceinline__ void block_sum_atomic(float v, float* red, int tid, float* dst) {
  red[tid] = v; __syncthreads();
  #pragma unroll
  for (int s = NT/2; s > 0; s >>= 1) {
    if (tid < s) red[tid] += red[tid + s];
    __syncthreads();
  }
  if (tid == 0)
    __hip_atomic_fetch_add(dst, red[0], __ATOMIC_RELAXED, __HIP_MEMORY_SCOPE_AGENT);
  __syncthreads();
}

// ---------------------------------------------------------------- init
__global__ void init_kernel(float* ws) {
  const int t = threadIdx.x;
  if (t < 5) ws[8 + t] = 0.f;
}

// ---------------------------------------------------------------- per-image stats
__global__ __launch_bounds__(256)
void stats_kernel(const float* __restrict__ ir, float* __restrict__ ws) {
  __shared__ float rs[256], rq[256];
  const int b = blockIdx.x, tid = threadIdx.x;
  const size_t base = (size_t)b * (H_*W_);
  float s = 0.f, q = 0.f;
  for (int i = tid; i < H_*W_; i += 256) {
    const float v = ir[base + i];
    s += v; q += v*v;
  }
  rs[tid] = s; rq[tid] = q; __syncthreads();
  for (int st = 128; st > 0; st >>= 1) {
    if (tid < st) { rs[tid] += rs[tid+st]; rq[tid] += rq[tid+st]; }
    __syncthreads();
  }
  if (tid == 0) {
    const float n = (float)(H_*W_);
    const float mean = rs[0] / n;
    float var = (rq[0] - n*mean*mean) / (n - 1.f);   // ddof=1
    var = fmaxf(var, 0.f);
    ws[b] = mean + 0.5f * (sqrtf(var) + 1e-6f);      // thr
  }
}

// ---------------------------------------------------------------- fused tile kernel
__global__ __launch_bounds__(NT)
void fusion_main_kernel(const float* __restrict__ ir,
                        const float* __restrict__ vis,
                        const float* __restrict__ fu,
                        float* __restrict__ ws)
{
  __shared__ float sIr[RT*RP], sVis[RT*RP], sFu[RT*RP];
  __shared__ __align__(16) float smem_scr[SCRF];  // phases 2-3: box temps; phase 4: f16 ssim scratch
  __shared__ float red[NT];

  float* tIr  = smem_scr;
  float* tVis = smem_scr + BR*BP;
  float* tFc  = smem_scr + 2*BR*BP;

  const int tid = threadIdx.x;
  const int tx = blockIdx.x, ty = blockIdx.y, b = blockIdx.z;
  const int x0 = tx*TILE, y0 = ty*TILE;
  const size_t base = (size_t)b * (H_*W_);
  const float thr = ws[b];

  // ---- Phase 1: cooperative halo load, zero-padded at image border
  for (int i = tid; i < RT*RT; i += NT) {
    const int r = i / RT, c = i - r*RT;
    const int gy = y0 + r - HALO, gx = x0 + c - HALO;
    const bool in = (gy >= 0) & (gy < H_) & (gx >= 0) & (gx < W_);
    const size_t g = base + (size_t)(in ? gy : 0) * W_ + (in ? gx : 0);
    sIr [r*RP + c] = in ? ir [g] : 0.f;
    sVis[r*RP + c] = in ? vis[g] : 0.f;
    sFu [r*RP + c] = in ? fu [g] : 0.f;
    if (in && gx + TILE < W_)              // warm L2 for the tile to the right
      __builtin_prefetch(&fu[g + TILE], 0, 1);
  }
  __syncthreads();

  // ---- Phase 2: 9x9 box blur, horizontal pass (rows -4..35 rel. tile)
  for (int i = tid; i < BR*TILE; i += NT) {
    const int r = i >> 5, c = i & (TILE-1);
    const int o = (r + 1)*RP + (c + 1);
    float si = 0.f, sv = 0.f, sf = 0.f;
    #pragma unroll
    for (int k = 0; k < 9; ++k) {
      si += sIr[o + k];
      sv += sVis[o + k];
      float f = sFu[o + k];
      sf += fminf(fmaxf(f, 0.f), 1.f);
    }
    tIr [r*BP + c] = si * (1.f/9.f);
    tVis[r*BP + c] = sv * (1.f/9.f);
    tFc [r*BP + c] = sf * (1.f/9.f);
  }
  __syncthreads();

  // ---- Phase 3: per-pixel terms (sobel, base, detail, bloom) -- rolled loop
  float accG = 0.f, accB = 0.f, accD = 0.f, accL = 0.f, accS = 0.f;
  for (int p = 0; p < (TILE*TILE)/NT; ++p) {
    const int idx = tid + p*NT;
    const int r = idx >> 5, c = idx & 31;

    const int sb = (r + 4)*RP + (c + 4);
    const float sobf = sobel_at(sFu,  sb);
    const float sobv = sobel_at(sVis, sb);
    const float sobi = sobel_at(sIr,  sb);
    accG += fabsf(sobf - fmaxf(sobv, sobi));

    float ib = 0.f, vb = 0.f, fb = 0.f;
    #pragma unroll
    for (int k = 0; k < 9; ++k) {
      const int o = (r + k)*BP + c;
      ib += tIr[o]; vb += tVis[o]; fb += tFc[o];
    }
    ib *= (1.f/9.f); vb *= (1.f/9.f); fb *= (1.f/9.f);

    const float w1 = 1.f / (1.f + __expf((vb - ib) * 5.f));   // softmax/tau=0.2
    const float tb = w1*ib + (1.f - w1)*vb;
    accB += fabsf(fb - tb);

    const float irv = sIr[(r + 5)*RP + (c + 5)];
    float fcv = sFu[(r + 5)*RP + (c + 5)];
    fcv = fminf(fmaxf(fcv, 0.f), 1.f);
    const float m = 1.f / (1.f + __expf(-20.f*(irv - thr)));
    accD += m * fabsf((fcv - fb) - (irv - ib));
    accL += m * fmaxf(fb - vb, 0.f);
  }
  __syncthreads();   // box temps dead; smem_scr reused as ssim f16 scratch below

  // ---- Phase 4: SSIM 11x11 gaussian, BOTH passes on the matrix pipe.
  // horizontal: tmpH(26x16) = In(26x26) x Gh(26x16 band)  -> 2 wmma (rows 16+10)
  // vertical:   map(16x16)  = Gv(16x32 band) x tmpH(26x16) -> 1 wmma
  // Both bands are zero for K>=26, so K-pad regions may hold (finite) garbage.
  const int lane = tid & 31;
  const int wid  = tid >> 5;
  const int oy = (wid >> 1) * 16, ox = (wid & 1) * 16;
  const int mrow = lane & 15;
  const int khi  = (lane >= 16) ? 8 : 0;
  const int rowoff = (lane >= 16) ? 8 : 0;     // D-layout row offset
  _Float16* hbase = reinterpret_cast<_Float16*>(smem_scr) + (size_t)wid * 5 * (16*KP);

  // Gh band (B operand, 32x16 f16): B[k,c] = g[k-c], 0<=k-c<=10
  // B layout: lanes 0-15 col=lane K=0..15; lanes 16-31 col=lane-16 K=16..31
  v16h ghB;
  {
    const int gc = lane & 15;
    const int gkb = (lane >= 16) ? 16 : 0;
    #pragma unroll
    for (int e = 0; e < 16; ++e) {
      const int d = (gkb + e) - gc;
      ghB[e] = (_Float16)((d >= 0 && d <= 10) ? G11[d] : 0.f);
    }
  }
  // Gv band (A operand, 16x32 f16): A[m,k] = g[k-m], 0<=k-m<=10
  // A layout: M=lane%16; lanes<16 hold K=0..7,16..23; lanes>=16 hold K=8..15,24..31
  v16h gvA;
  #pragma unroll
  for (int h = 0; h < 16; ++h) {
    const int k = (h < 8) ? (h + khi) : (h + 8 + khi);
    const int d = k - mrow;
    gvA[h] = (_Float16)((d >= 0 && d <= 10) ? G11[d] : 0.f);
  }

  // horizontal pass: raw F/V loaded once per row-half, shared by all 5 channels
  #pragma unroll
  for (int half = 0; half < 2; ++half) {
    const int rowbase = half * 16;
    float Fc[16], Vv[16];
    #pragma unroll
    for (int e = 0; e < 16; ++e) {
      const int k = e + 2*khi;
      const int rr = rowbase + mrow;
      const bool ok = (rr < 26) & (k < 26);
      const int o = (oy + rr)*RP + (ox + k);
      const float f = ok ? sFu[o]  : 0.f;
      Vv[e]         = ok ? sVis[o] : 0.f;
      Fc[e] = fminf(fmaxf(f, 0.f), 1.f);
    }
    #pragma unroll
    for (int ch = 0; ch < 5; ++ch) {
      v16h aIn;
      #pragma unroll
      for (int e = 0; e < 16; ++e) {
        const float f = Fc[e], v = Vv[e];
        const float val = (ch==0) ? f : (ch==1) ? v : (ch==2) ? f*f : (ch==3) ? v*v : f*v;
        aIn[e] = (_Float16)val;
      }
      v8f cz = {};
      const v8f D = __builtin_amdgcn_wmma_f32_16x16x32_f16(
          false, aIn, false, ghB, (short)0, cz, false, false);
      // store tmpH rows to scratch in vertical-B layout: half pairs at [c*KP + j]
      _Float16* hW = hbase + ch * (16*KP);
      const int ccol = lane & 15;
      #pragma unroll
      for (int i = 0; i < 8; i += 2) {
        const int j = rowbase + i + rowoff;
        if (j < 26) {
          v2h p; p[0] = (_Float16)D[i]; p[1] = (_Float16)D[i+1];
          *reinterpret_cast<v2h*>(&hW[ccol*KP + j]) = p;
        }
      }
    }
  }
  __builtin_amdgcn_wave_barrier();

  // vertical pass: B loaded as two 16B vectors (ds_load_b128) per lane
  v8f res[5];
  #pragma unroll
  for (int ch = 0; ch < 5; ++ch) {
    const _Float16* hR = hbase + ch * (16*KP);
    const int ccol = lane & 15;
    const int kb   = (lane >= 16) ? 8 : 0;
    const v8h lo = *reinterpret_cast<const v8h*>(&hR[ccol*KP + kb]);
    const v8h hi = *reinterpret_cast<const v8h*>(&hR[ccol*KP + kb + 16]);
    const v16h bMat = __builtin_shufflevector(lo, hi, 0,1,2,3,4,5,6,7,8,9,10,11,12,13,14,15);
    v8f cz = {};
    res[ch] = __builtin_amdgcn_wmma_f32_16x16x32_f16(
        false, gvA, false, bMat, (short)0, cz, false, false);
  }

  // pointwise SSIM map; valid-padding mask (window centers 5..506)
  const float C1c = 1e-4f, C2c = 9e-4f;
  #pragma unroll
  for (int i = 0; i < 8; ++i) {
    const int M = i + rowoff;
    const int N = lane & 15;
    const int gy = y0 + oy + M, gx = x0 + ox + N;
    const float mu1 = res[0][i], mu2 = res[1][i];
    const float s1  = res[2][i] - mu1*mu1;
    const float s2  = res[3][i] - mu2*mu2;
    const float s12 = res[4][i] - mu1*mu2;
    const float num = (2.f*mu1*mu2 + C1c) * (2.f*s12 + C2c);
    const float den = (mu1*mu1 + mu2*mu2 + C1c) * (s1 + s2 + C2c);
    const bool valid = (gy >= HALO) & (gy < H_-HALO) & (gx >= HALO) & (gx < W_-HALO);
    accS += valid ? (num/den) : 0.f;
  }

  // ---- Phase 5: block reduce + global accumulate
  block_sum_atomic(accG, red, tid, &ws[8]);
  block_sum_atomic(accB, red, tid, &ws[9]);
  block_sum_atomic(accD, red, tid, &ws[10]);
  block_sum_atomic(accL, red, tid, &ws[11]);
  block_sum_atomic(accS, red, tid, &ws[12]);
}

// ---------------------------------------------------------------- finalize
__global__ void finalize_kernel(const float* __restrict__ ws, float* __restrict__ out) {
  const float N  = (float)B_ * H_ * W_;
  const float Ns = (float)B_ * (H_-10) * (W_-10);
  const float loss_grad = ws[8]  / N;
  const float loss_base = ws[9]  / N;
  const float loss_det  = ws[10] / N;
  const float loss_blm  = ws[11] / N;
  const float loss_ssim = 1.f - ws[12] / Ns;
  out[0] = loss_base + 3.f*loss_det + 2.f*loss_grad + 0.5f*loss_ssim + 0.2f*loss_blm;
}

extern "C" void kernel_launch(void* const* d_in, const int* in_sizes, int n_in,
                              void* d_out, int out_size, void* d_ws, size_t ws_size,
                              hipStream_t stream) {
  (void)in_sizes; (void)n_in; (void)out_size; (void)ws_size;
  const float* ir  = (const float*)d_in[0];
  const float* vis = (const float*)d_in[1];
  const float* fu  = (const float*)d_in[2];
  float* ws  = (float*)d_ws;
  float* out = (float*)d_out;

  hipLaunchKernelGGL(init_kernel,  dim3(1),  dim3(32), 0, stream, ws);
  hipLaunchKernelGGL(stats_kernel, dim3(B_), dim3(256), 0, stream, ir, ws);
  hipLaunchKernelGGL(fusion_main_kernel,
                     dim3(W_/TILE, H_/TILE, B_), dim3(NT), 0, stream,
                     ir, vis, fu, ws);
  hipLaunchKernelGGL(finalize_kernel, dim3(1), dim3(1), 0, stream, ws, out);
}
